// AttentionOT_29626684408322
// MI455X (gfx1250) — matrix operand order repaired and checked by reading, not compiled, not run on hardware
//
#include <hip/hip_runtime.h>
#include <cmath>

typedef __bf16 bf16;
typedef __attribute__((ext_vector_type(16))) __bf16 v16bf;
typedef __attribute__((ext_vector_type(8)))  __bf16 bf16x8;
typedef __attribute__((ext_vector_type(8)))  float   v8f;
typedef __attribute__((ext_vector_type(4)))  unsigned int u32x4;
typedef __attribute__((ext_vector_type(8)))  int    i32x8;
typedef __attribute__((ext_vector_type(4)))  int    i32x4;
typedef unsigned short u16t;

#define NQ 1024
#define BB 8
#define MM 1024
#define CC 1024

__device__ __forceinline__ u16t f2bf(float x){
  unsigned u = __builtin_bit_cast(unsigned, x);
  u += 0x7FFFu + ((u >> 16) & 1u);       // round-to-nearest-even
  return (u16t)(u >> 16);
}

// ---------------------------------------------------------------- conversions
__global__ __launch_bounds__(256) void cvt_f32_bf16(const float* __restrict__ in,
                                                    u16t* __restrict__ out, int n){
  int i = blockIdx.x * 256 + threadIdx.x;
  if (i < n) out[i] = f2bf(in[i]);
}

// W [K,N] f32 -> Wt [N,K] bf16 (LDS-tiled 32x32 transpose)
__global__ __launch_bounds__(256) void cvt_tr_bf16(const float* __restrict__ in,
                                                   u16t* __restrict__ out){
  __shared__ float t[32][33];
  int tx = threadIdx.x & 31, ty = threadIdx.x >> 5;
  int k0 = blockIdx.y * 32, n0 = blockIdx.x * 32;
  #pragma unroll
  for (int i = 0; i < 4; i++)
    t[ty + 8 * i][tx] = in[(size_t)(k0 + ty + 8 * i) * CC + n0 + tx];
  __syncthreads();
  #pragma unroll
  for (int i = 0; i < 4; i++)
    out[(size_t)(n0 + ty + 8 * i) * CC + k0 + tx] = f2bf(t[tx][ty + 8 * i]);
}

__global__ __launch_bounds__(256) void fillz(float* __restrict__ p, int n){
  int i = blockIdx.x * 256 + threadIdx.x;
  if (i < n) p[i] = 0.0f;
}

// ---------------------------------------------------------------- TDM helper
// One tensor_load_to_lds: copy `rows` rows x 32 bf16 from global (row stride
// `stride` elems) into LDS at byte addr `ldsb`, padding 16B after every 64B
// so LDS rows land on an 80B (LDSA=40 bf16) stride.
__device__ __forceinline__ void tdm_tile_2d(unsigned ldsb, const u16t* gptr,
                                            unsigned rows, unsigned stride){
  unsigned long long ga = (unsigned long long)(size_t)gptr;
  u32x4 g0;
  g0[0] = 1u;                                          // count=1, user mode
  g0[1] = ldsb;                                        // lds_addr
  g0[2] = (unsigned)(ga & 0xFFFFFFFFu);                // global_addr[31:0]
  g0[3] = (unsigned)((ga >> 32) & 0x1FFFFFFu)          // global_addr[56:32]
        | (2u << 30);                                  // type=2 (image)
  i32x8 g1;
  g1[0] = (int)((1u << 16)                             // data_size = 2B
              | (1u << 20)                             // pad_enable
              | (3u << 22)                             // pad_interval: 16 DW
              | (3u << 25));                           // pad_amount: 4 DW
  const unsigned td0 = 0x7FFFFFFFu, td1 = 0x7FFFFFFFu; // no OOB clipping
  g1[1] = (int)((td0 & 0xFFFFu) << 16);                // tensor_dim0 lo
  g1[2] = (int)((td0 >> 16) | ((td1 & 0xFFFFu) << 16));// td0 hi | td1 lo
  g1[3] = (int)((td1 >> 16) | (32u << 16));            // td1 hi | tile_dim0=32
  g1[4] = (int)(rows & 0xFFFFu);                       // tile_dim1 ; tile_dim2=0
  g1[5] = (int)stride;                                 // dim0_stride[31:0]
  g1[6] = 0;                                           // dim0_stride hi | dim1_stride lo
  g1[7] = 0;
  i32x4 gz4 = {0, 0, 0, 0};
  i32x8 gz8 = {0, 0, 0, 0, 0, 0, 0, 0};
  __builtin_amdgcn_tensor_load_to_lds(g0, g1, gz4, gz4, gz8, 0);
}

// ---------------------------------------------------------------- bf16 GEMM
// C[M,N] = A[M,K] * B^T where B is stored [N,K] row-major (all operands are
// K-major). TDM double-buffered LDS staging, v_wmma_f32_16x16x32_bf16 compute.
// OM: 0 = f32 out, 1 = f32 + bias, 2 = bf16 out, 3 = bf16 transposed V store.
#define TBM 128
#define TBN 64
#define TBK 32
#define LDSA 40   // padded bf16 row stride (80B: 16B-aligned, bank-spread)

template<int OM>
__global__ __launch_bounds__(256) void gemm_bf16(
    const u16t* __restrict__ Ag, int lda, long long sA,
    const u16t* __restrict__ Bg, int ldb, long long sB,
    float* __restrict__ Cf, u16t* __restrict__ Cb, int ldc, long long sC,
    const float* __restrict__ bias, int K)
{
  __shared__ __align__(16) u16t As[2][TBM * LDSA];
  __shared__ __align__(16) u16t Bs[2][TBN * LDSA];

  const int tid   = threadIdx.x;
  const int lane  = tid & 31;
  const int wave  = tid >> 5;
  const int wm    = wave >> 1;     // 0..3
  const int wn    = wave & 1;      // 0..1
  const int lmod  = lane & 15;
  const int lhalf = lane >> 4;

  const long long bz = blockIdx.z;
  Ag += bz * sA;  Bg += bz * sB;
  const size_t rowBase = (size_t)blockIdx.y * TBM;
  const size_t colBase = (size_t)blockIdx.x * TBN;
  const u16t* Atile = Ag + rowBase * (size_t)lda;
  const u16t* Btile = Bg + colBase * (size_t)ldb;

  v8f acc[2][2] = {};
  const int nt = K / TBK;

  if (tid < 32){  // wave 0 drives the Tensor Data Mover
    tdm_tile_2d((unsigned)(size_t)(void*)&As[0][0], Atile, TBM, (unsigned)lda);
    tdm_tile_2d((unsigned)(size_t)(void*)&Bs[0][0], Btile, TBN, (unsigned)ldb);
  }

  for (int t = 0; t < nt; ++t){
    const int cur = t & 1;
    if (tid < 32){
      if (t + 1 < nt){
        tdm_tile_2d((unsigned)(size_t)(void*)&As[cur ^ 1][0],
                    Atile + (t + 1) * TBK, TBM, (unsigned)lda);
        tdm_tile_2d((unsigned)(size_t)(void*)&Bs[cur ^ 1][0],
                    Btile + (t + 1) * TBK, TBN, (unsigned)ldb);
        __builtin_amdgcn_s_wait_tensorcnt(2);   // tile t (2 oldest ops) done
      } else {
        __builtin_amdgcn_s_wait_tensorcnt(0);
      }
    }
    __syncthreads();                            // publish LDS tile to all waves

    const u16t* Ac = As[cur];
    const u16t* Bc = Bs[cur];
    v16bf afr[2], bfr[2];
    #pragma unroll
    for (int mi = 0; mi < 2; mi++){             // A: lanes0-15 K{0-7,16-23}
      int r = wm * 32 + mi * 16 + lmod;
      const bf16x8* pa = (const bf16x8*)(Ac + r * LDSA + lhalf * 8);
      bf16x8 a0 = pa[0], a1 = pa[2];
      #pragma unroll
      for (int e = 0; e < 8; e++){ afr[mi][e] = a0[e]; afr[mi][8 + e] = a1[e]; }
    }
    #pragma unroll
    for (int ni = 0; ni < 2; ni++){             // B: lanes0-15 K{0-7,16-23}
      int c = wn * 32 + ni * 16 + lmod;
      const bf16x8* pb = (const bf16x8*)(Bc + c * LDSA + lhalf * 8);
      bf16x8 b0 = pb[0], b1 = pb[2];
      #pragma unroll
      for (int e = 0; e < 8; e++){ bfr[ni][e] = b0[e]; bfr[ni][8 + e] = b1[e]; }
    }
    #pragma unroll
    for (int mi = 0; mi < 2; mi++)
      #pragma unroll
      for (int ni = 0; ni < 2; ni++)
        acc[mi][ni] = __builtin_amdgcn_wmma_f32_16x16x32_bf16(
            false, afr[mi], false, bfr[ni], (short)0, acc[mi][ni], false, false);
    __syncthreads();                            // LDS reusable for next TDM
  }

  // ---- epilogue: lanes 0-15 -> M=r, lanes 16-31 -> M=8+r, N=lmod
  #pragma unroll
  for (int mi = 0; mi < 2; mi++){
    #pragma unroll
    for (int ni = 0; ni < 2; ni++){
      int col = (int)colBase + wn * 32 + ni * 16 + lmod;
      float badd = (OM == 1) ? bias[col] : 0.0f;
      #pragma unroll
      for (int r = 0; r < 8; r++){
        size_t row = rowBase + wm * 32 + mi * 16 + r + 8 * lhalf;
        if (OM == 3){
          // Vt[b, col, m] = acc ; row = b*1024 + m
          size_t idx = (((row >> 10) << 10) + col) * 1024 + (row & 1023);
          Cb[idx] = f2bf(acc[mi][ni][r]);
        } else {
          size_t idx = row * (size_t)ldc + col + (size_t)(bz * sC);
          if (OM == 2)      Cb[idx] = f2bf(acc[mi][ni][r]);
          else              Cf[idx] = acc[mi][ni][r] + badd;
        }
      }
    }
  }
}

// ---------------------------------------------------------------- L2 normalize
template<bool PERM>
__global__ __launch_bounds__(256) void l2norm_bf16(const float* __restrict__ in,
                                                   u16t* __restrict__ out){
  __shared__ float red[8];
  const int row = blockIdx.x;
  const float* r = in + (size_t)row * CC;
  float ss = 0.0f;
  for (int i = threadIdx.x; i < CC; i += 256){ float x = r[i]; ss += x * x; }
  #pragma unroll
  for (int o = 16; o; o >>= 1) ss += __shfl_xor(ss, o, 32);
  if ((threadIdx.x & 31) == 0) red[threadIdx.x >> 5] = ss;
  __syncthreads();
  float tot = red[0]+red[1]+red[2]+red[3]+red[4]+red[5]+red[6]+red[7];
  float inv = 1.0f / fmaxf(sqrtf(tot), 1e-12f);
  size_t obase;
  if (PERM){ int n = row >> 3, b = row & 7; obase = ((size_t)b * NQ + n) * CC; }
  else       obase = (size_t)row * CC;
  for (int i = threadIdx.x; i < CC; i += 256) out[obase + i] = f2bf(r[i] * inv);
}

// ---------------------------------------------------------------- Sinkhorn
__device__ __forceinline__ void lse_step(float& m, float& s, float a){
  float m2 = fmaxf(m, a);
  s = s * __expf(m - m2) + __expf(a - m2);
  m = m2;
}
__device__ __forceinline__ void lse_wave_reduce(float& m, float& s){
  #pragma unroll
  for (int o = 16; o; o >>= 1){
    float mo = __shfl_xor(m, o, 32), so = __shfl_xor(s, o, 32);
    float m2 = fmaxf(m, mo);
    s = s * __expf(m - m2) + so * __expf(mo - m2);
    m = m2;
  }
}

__global__ __launch_bounds__(256) void sink_u(const float* __restrict__ sim,
                                              float* __restrict__ u,
                                              const float* __restrict__ vd,
                                              float inv_eps, float eps, float log_mu){
  int row  = blockIdx.x * 8 + (threadIdx.x >> 5);   // b*MM+m
  int lane = threadIdx.x & 31;
  int b    = row >> 10;
  const float* srow = sim + (size_t)row * NQ;
  const float* vrow = vd + b * NQ;
  float uo = u[row];
  float m = -3.0e38f, s = 0.0f;
  for (int j = lane; j < NQ; j += 32)
    lse_step(m, s, (srow[j] - 1.0f + uo + vrow[j]) * inv_eps);
  lse_wave_reduce(m, s);
  float lse = m + __logf(s);
  if (lane == 0) u[row] = eps * (log_mu - lse) + uo;
}

__global__ __launch_bounds__(256) void sink_v(const float* __restrict__ sim,
                                              const float* __restrict__ u,
                                              float* __restrict__ vd,
                                              float inv_eps, float eps, float log_nu){
  __shared__ float su[MM];
  int b = blockIdx.x >> 2;
  int n = ((blockIdx.x & 3) << 8) + threadIdx.x;
  for (int i = threadIdx.x; i < MM; i += 256) su[i] = u[b * MM + i];
  __syncthreads();
  float vo = vd[b * NQ + n];
  const float* sb = sim + (size_t)b * MM * NQ;
  float m = -3.0e38f, s = 0.0f;
  for (int mm = 0; mm < MM; mm++)
    lse_step(m, s, (sb[(size_t)mm * NQ + n] - 1.0f + su[mm] + vo) * inv_eps);
  vd[b * NQ + n] = eps * (log_nu - (m + __logf(s))) + vo;
}

__global__ __launch_bounds__(256) void attn_row(const float* __restrict__ sim,
                                                const float* __restrict__ u,
                                                const float* __restrict__ vd,
                                                float inv_eps, float* __restrict__ attn){
  int row  = blockIdx.x * 8 + (threadIdx.x >> 5);
  int lane = threadIdx.x & 31;
  int b    = row >> 10;
  const float* srow = sim + (size_t)row * NQ;
  const float* vrow = vd + b * NQ;
  float uo = u[row];
  float acc = 0.0f;
  for (int j = lane; j < NQ; j += 32){
    float sv = srow[j];
    acc += sv * __expf((sv - 1.0f + uo + vrow[j]) * inv_eps);
  }
  #pragma unroll
  for (int o = 16; o; o >>= 1) acc += __shfl_xor(acc, o, 32);
  if (lane == 0) attn[row] = 1048576.0f * acc;   // M*Nq
}

__global__ __launch_bounds__(256) void make_Tt(const float* __restrict__ sim,
                                               const float* __restrict__ u,
                                               const float* __restrict__ vd,
                                               u16t* __restrict__ Tt, float inv_eps){
  __shared__ u16t t[32][33];
  int b  = blockIdx.z;
  int tN = blockIdx.x * 32, tM = blockIdx.y * 32;
  int tx = threadIdx.x & 31, ty = threadIdx.x >> 5;
  #pragma unroll
  for (int i = 0; i < 4; i++){
    int m = tM + ty + 8 * i, n = tN + tx;
    float T = __expf((sim[((size_t)b * MM + m) * NQ + n] - 1.0f
                      + u[b * MM + m] + vd[b * NQ + n]) * inv_eps);
    t[tx][ty + 8 * i] = f2bf(T);
  }
  __syncthreads();
  #pragma unroll
  for (int i = 0; i < 4; i++){
    int n = tN + ty + 8 * i, m = tM + tx;
    Tt[((size_t)b * NQ + n) * MM + m] = t[ty + 8 * i][tx];
  }
}

// ---------------------------------------------------------------- host
extern "C" void kernel_launch(void* const* d_in, const int* in_sizes, int n_in,
                              void* d_out, int out_size, void* d_ws, size_t ws_size,
                              hipStream_t stream){
  (void)in_sizes; (void)n_in; (void)out_size; (void)ws_size;
  const float* xq = (const float*)d_in[0];
  const float* xk = (const float*)d_in[1];
  const float* xv = (const float*)d_in[2];
  const float* Wq = (const float*)d_in[3];
  const float* Wk = (const float*)d_in[4];
  const float* Wv = (const float*)d_in[5];
  const float* Wp = (const float*)d_in[6];
  const float* bp = (const float*)d_in[7];

  float* outx    = (float*)d_out;                       // [Nq,B,C]
  float* outattn = outx + (size_t)NQ * BB * CC;         // [B,M]

  char* base = (char*)d_ws;  size_t off = 0;
  auto alloc = [&](size_t bytes)->void*{
    void* p = base + off; off += (bytes + 255) & ~(size_t)255; return p; };

  const size_t R = (size_t)NQ * BB;                     // 8192 rows
  u16t* A1  = (u16t*)alloc(R * CC * 2);  // xq_bf  -> later Vt_bf [B,C,M]
  u16t* A2  = (u16t*)alloc(R * CC * 2);  // xk_bf  -> later Qn_bf [B,Nq,C] -> Tt_bf
  u16t* A3  = (u16t*)alloc(R * CC * 2);  // xv_bf  -> later Kn_bf [B,M,C]  -> x_bf
  u16t* Wqt = (u16t*)alloc((size_t)CC * CC * 2);   // transposed bf16 weights
  u16t* Wkt = (u16t*)alloc((size_t)CC * CC * 2);
  u16t* Wvt = (u16t*)alloc((size_t)CC * CC * 2);
  u16t* Wpt = (u16t*)alloc((size_t)CC * CC * 2);
  float* F1 = (float*)alloc(R * CC * 4); // qbuf -> sim [B,M,Nq]
  float* F2 = (float*)alloc(R * CC * 4); // kbuf
  float* U  = (float*)alloc((size_t)BB * MM * 4);
  float* Vd = (float*)alloc((size_t)BB * NQ * 4);

  const float eps = 0.05f, inv_eps = 1.0f / 0.05f;
  const float log_mu = logf(1.0f / MM + 1e-8f);
  const float log_nu = logf(1.0f / NQ + 1e-8f);
  const dim3 blk(256);
  const int nA = (int)(R * CC);

  cvt_f32_bf16<<<nA / 256, blk, 0, stream>>>(xq, A1, nA);
  cvt_f32_bf16<<<nA / 256, blk, 0, stream>>>(xk, A2, nA);
  cvt_f32_bf16<<<nA / 256, blk, 0, stream>>>(xv, A3, nA);
  cvt_tr_bf16<<<dim3(32, 32), blk, 0, stream>>>(Wq, Wqt);
  cvt_tr_bf16<<<dim3(32, 32), blk, 0, stream>>>(Wk, Wkt);
  cvt_tr_bf16<<<dim3(32, 32), blk, 0, stream>>>(Wv, Wvt);
  cvt_tr_bf16<<<dim3(32, 32), blk, 0, stream>>>(Wp, Wpt);

  // q = xq@Wq (f32), k = xk@Wk (f32), v = xv@Wv -> bf16 transposed Vt into A1
  gemm_bf16<0><<<dim3(CC/TBN, R/TBM, 1), blk, 0, stream>>>(
      A1, CC, 0, Wqt, CC, 0, F1, nullptr, CC, 0, nullptr, CC);
  gemm_bf16<0><<<dim3(CC/TBN, R/TBM, 1), blk, 0, stream>>>(
      A2, CC, 0, Wkt, CC, 0, F2, nullptr, CC, 0, nullptr, CC);
  gemm_bf16<3><<<dim3(CC/TBN, R/TBM, 1), blk, 0, stream>>>(
      A3, CC, 0, Wvt, CC, 0, nullptr, A1, 0, 0, nullptr, CC);

  // normalize: Qn -> [B,Nq,C] (A2), Kn -> [B,M,C] (A3)
  l2norm_bf16<true ><<<(int)R, blk, 0, stream>>>(F1, A2);
  l2norm_bf16<false><<<(int)R, blk, 0, stream>>>(F2, A3);

  // sim[b] = Kn_b (M,C) x Qn_b^T  -> F1 [B,M,Nq]
  gemm_bf16<0><<<dim3(NQ/TBN, MM/TBM, BB), blk, 0, stream>>>(
      A3, CC, (long long)MM * CC, A2, CC, (long long)NQ * CC,
      F1, nullptr, NQ, (long long)MM * NQ, nullptr, CC);

  // Sinkhorn (33.5 MB cost matrix stays L2-resident on MI455X)
  fillz<<<(BB * MM) / 256, blk, 0, stream>>>(U,  BB * MM);
  fillz<<<(BB * NQ) / 256, blk, 0, stream>>>(Vd, BB * NQ);
  for (int it = 0; it < 100; ++it){
    sink_u<<<BB * MM / 8,   blk, 0, stream>>>(F1, U, Vd, inv_eps, eps, log_mu);
    sink_v<<<BB * NQ / 256, blk, 0, stream>>>(F1, U, Vd, inv_eps, eps, log_nu);
  }

  // attn_save and transposed transport plan
  attn_row<<<BB * MM / 8, blk, 0, stream>>>(F1, U, Vd, inv_eps, outattn);
  make_Tt<<<dim3(NQ/32, MM/32, BB), blk, 0, stream>>>(F1, U, Vd, A2, inv_eps);

  // x[n,b,:] = T_b^T (Nq,M) x Vt_b^T ; Vt_b is [C,M] = [N,K] -> bf16 into A3
  gemm_bf16<2><<<dim3(CC/TBN, NQ/TBM, BB), blk, 0, stream>>>(
      A2, MM, (long long)NQ * MM, A1, MM, (long long)CC * MM,
      nullptr, A3, BB * CC, (long long)CC, nullptr, MM);

  // out = x @ Wp + bp (f32 -> d_out)
  gemm_bf16<1><<<dim3(CC/TBN, R/TBM, 1), blk, 0, stream>>>(
      A3, CC, 0, Wpt, CC, 0, outx, nullptr, CC, 0, bp, CC);
}
